// pugeonet_65017214927177
// MI455X (gfx1250) — matrix operand order repaired and checked by reading, not compiled, not run on hardware
//
#include <hip/hip_runtime.h>

typedef unsigned short bhalf;                                   // raw bf16 bits
typedef __attribute__((ext_vector_type(16))) __bf16 v16bf;
typedef __attribute__((ext_vector_type(8)))  float  v8f;

#define NPTS 4096
#define NB 4

__device__ __forceinline__ bhalf f2b(float f) {
  unsigned int u = __float_as_uint(f);
  unsigned int r = u + 0x7FFFu + ((u >> 16) & 1u);
  return (bhalf)(r >> 16);
}
__device__ __forceinline__ float b2f(bhalf h) {
  return __uint_as_float(((unsigned int)h) << 16);
}
__device__ __forceinline__ float lrelu(float v) { return v >= 0.0f ? v : 0.2f * v; }

// pack two bf16 rows (even c, odd c) into pair-interleaved dwords:
// o0 = {odd.m0, even.m0}, o1 = {odd.m1, even.m1}
__device__ __forceinline__ void pack2(unsigned int ev, unsigned int od,
                                      unsigned int& o0, unsigned int& o1) {
  o0 = (ev & 0xFFFFu) | (od << 16);
  o1 = (ev >> 16) | (od & 0xFFFF0000u);
}

// ---------------------------------------------------------------------------
// Fold BN into conv weights: W' = W * g/sqrt(1+eps), b' = b*g/sqrt(1+eps)+be
// ---------------------------------------------------------------------------
__global__ __launch_bounds__(256) void fold_params(
    const float* __restrict__ W, const float* __restrict__ b,
    const float* __restrict__ g, const float* __restrict__ be,
    int O, int C, int has_bn,
    bhalf* __restrict__ Wb, float* __restrict__ Wf, float* __restrict__ bf) {
  int i = blockIdx.x * 256 + threadIdx.x;
  float invs = rsqrtf(1.0f + 1e-5f);
  if (i < O * C) {
    int o = i / C;
    float s = has_bn ? g[o] * invs : 1.0f;
    float w = W[i] * s;
    Wb[i] = f2b(w);
    Wf[i] = w;
  }
  if (i < O) {
    float s = has_bn ? g[i] * invs : 1.0f;
    bf[i] = b[i] * s + (has_bn ? be[i] : 0.0f);
  }
}

// ---------------------------------------------------------------------------
// k-NN on xyz (C=3), direct distance scan, top-15 per point
// ---------------------------------------------------------------------------
__global__ __launch_bounds__(256) void knn_xyz_k15(const float* __restrict__ x,
                                                   int* __restrict__ idxOut) {
  const int N = NPTS, KS = 15;
  int b = blockIdx.z;
  int n = blockIdx.x * 256 + threadIdx.x;
  const float* xb = x + (long)b * 6 * N;
  float cx = xb[n], cy = xb[N + n], cz = xb[2 * N + n];
  float cs = cx * cx + cy * cy + cz * cz;
  float bv[KS]; int bi[KS];
#pragma unroll
  for (int q = 0; q < KS; ++q) { bv[q] = -3.0e38f; bi[q] = 0; }
  __shared__ float tx[256], ty[256], tz[256], ts[256];
  for (int m0 = 0; m0 < N; m0 += 256) {
    __syncthreads();
    int m = m0 + threadIdx.x;
    float qx = xb[m], qy = xb[N + m], qz = xb[2 * N + m];
    tx[threadIdx.x] = qx; ty[threadIdx.x] = qy; tz[threadIdx.x] = qz;
    ts[threadIdx.x] = qx * qx + qy * qy + qz * qz;
    __syncthreads();
    for (int i = 0; i < 256; ++i) {
      float d = 2.0f * (cx * tx[i] + cy * ty[i] + cz * tz[i]) - cs - ts[i];
      if (d > bv[KS - 1]) {
        float cv = d; int ci = m0 + i;
#pragma unroll
        for (int q = 0; q < KS; ++q)
          if (cv > bv[q]) { float tv = bv[q]; int ti = bi[q]; bv[q] = cv; bi[q] = ci; cv = tv; ci = ti; }
      }
    }
  }
  int* op = idxOut + ((long)b * N + n) * KS;
#pragma unroll
  for (int q = 0; q < KS; ++q) op[q] = bi[q];
}

// ---------------------------------------------------------------------------
// Fused edge-feature + d1 conv (6 -> 128) + folded BN + leaky relu, bf16 out
// out layout: [b][o][n*15+k], M = N*15
// ---------------------------------------------------------------------------
__global__ __launch_bounds__(256) void edge_conv_d1(
    const float* __restrict__ x, const int* __restrict__ idx,
    const float* __restrict__ W, const float* __restrict__ bias,
    bhalf* __restrict__ out) {
  __shared__ float Ws[128 * 6];
  __shared__ float Bi[128];
  const int N = NPTS, Kn = 15, M = NPTS * 15;
  for (int i = threadIdx.x; i < 768; i += 256) Ws[i] = W[i];
  for (int i = threadIdx.x; i < 128; i += 256) Bi[i] = bias[i];
  __syncthreads();
  int e = blockIdx.x * 256 + threadIdx.x;
  int b = blockIdx.z;
  int n = e / Kn, k = e - n * Kn;
  const float* xb = x + (long)b * 6 * N;
  int j = idx[((long)b * N + n) * Kn + k];
  float cx = xb[n], cy = xb[N + n], cz = xb[2 * N + n];
  float f0 = xb[j] - cx, f1 = xb[N + j] - cy, f2 = xb[2 * N + j] - cz;
#pragma unroll 4
  for (int o = 0; o < 128; ++o) {
    const float* w = &Ws[o * 6];
    float a = Bi[o] + w[0] * f0 + w[1] * f1 + w[2] * f2 + w[3] * cx + w[4] * cy + w[5] * cz;
    out[((long)(b * 128 + o)) * M + e] = f2b(lrelu(a));
  }
}

// ---------------------------------------------------------------------------
// Generic bf16 WMMA GEMM: Y[b][o][m] = act(W[o,:] . X[b][:, m] + bias[o])
// Block: 256 thr = 8 waves; tile 64(O) x 64(M); K-step 32.
// A staged [o][c] (b128 fragment loads); B staged pair-packed [kpair][m]
// so each B fragment dword is a single ds_load_b32.
// Requires O%64==0, M%64==0, C%32==0.
// ---------------------------------------------------------------------------
__global__ __launch_bounds__(256) void gemm_bf16(
    const bhalf* __restrict__ W, const float* __restrict__ bias,
    const bhalf* __restrict__ X, long xBatch,
    bhalf* __restrict__ Y, long yBatch,
    int O, int C, int M, int act) {
  __shared__ __align__(16) bhalf As[64 * 32];          // [o][c]
  __shared__ __align__(16) unsigned int Bs2[16 * 64];  // [kpair][m] pair-packed
  const int tid = threadIdx.x;
  const int wave = tid >> 5;
  const int lane = tid & 31;
  const int b = blockIdx.z;
  const int blockO = blockIdx.y * 64;
  const int blockM = blockIdx.x * 64;
  const bhalf* Xb = X + (long)b * xBatch;
  bhalf* Yb = Y + (long)b * yBatch;

  const int orow = (wave & 3) * 16;
  const int mbase = (wave >> 2) * 32;
  const int r = lane & 15;
  const int half = lane >> 4;

  const int aoRow = tid >> 2;           // 0..63
  const int aCol = (tid & 3) * 8;       // 0..24
  const int kp = tid >> 4;              // 0..15 (k-pair)
  const int pm = (tid & 15) * 4;        // 0..60 (m chunk of 4)

  v8f acc0 = {0, 0, 0, 0, 0, 0, 0, 0};
  v8f acc1 = {0, 0, 0, 0, 0, 0, 0, 0};

  for (int k0 = 0; k0 < C; k0 += 32) {
    __syncthreads();
    *reinterpret_cast<uint4*>(&As[aoRow * 32 + aCol]) =
        *reinterpret_cast<const uint4*>(W + (long)(blockO + aoRow) * C + k0 + aCol);
    {
      uint2 ev = *reinterpret_cast<const uint2*>(Xb + (long)(k0 + 2 * kp) * M + blockM + pm);
      uint2 od = *reinterpret_cast<const uint2*>(Xb + (long)(k0 + 2 * kp + 1) * M + blockM + pm);
      uint4 o;
      pack2(ev.x, od.x, o.x, o.y);
      pack2(ev.y, od.y, o.z, o.w);
      *reinterpret_cast<uint4*>(&Bs2[kp * 64 + pm]) = o;
    }
    if (k0 + 32 < C)
      __builtin_prefetch(Xb + (long)(k0 + 32 + 2 * kp) * M + blockM + pm, 0, 1);
    __syncthreads();

    union { unsigned int u[8]; v16bf v; } fa, fb0, fb1;
    const unsigned int* As32 = reinterpret_cast<const unsigned int*>(As);
#pragma unroll
    for (int j = 0; j < 8; ++j) {
      int kk = 2 * j + half * 16;   // A: contiguous k = 16*half + 2j, 2j+1
      fa.u[j] = As32[(orow + r) * 16 + (kk >> 1)];
      int kp2 = half * 8 + j;                                   // B: same contiguous k pairs as A
      fb0.u[j] = Bs2[kp2 * 64 + mbase + r];
      fb1.u[j] = Bs2[kp2 * 64 + mbase + 16 + r];
    }
    acc0 = __builtin_amdgcn_wmma_f32_16x16x32_bf16(false, fa.v, false, fb0.v, (short)0, acc0, false, false);
    acc1 = __builtin_amdgcn_wmma_f32_16x16x32_bf16(false, fa.v, false, fb1.v, (short)0, acc1, false, false);
  }

#pragma unroll
  for (int j = 0; j < 8; ++j) {
    int o = blockO + orow + j + 8 * half;
    float bv = bias[o];
    float v0 = acc0[j] + bv;
    float v1 = acc1[j] + bv;
    if (act) { v0 = lrelu(v0); v1 = lrelu(v1); }
    long row = (long)o * M + blockM + mbase + r;
    Yb[row] = f2b(v0);
    Yb[row + 16] = f2b(v1);
  }
}

// ---------------------------------------------------------------------------
// Gram / pairwise-distance tile kernel: pd[n][m] = 2*X^T X - xx[n] - xx[m]
// X: [C x N] bf16 view (per-batch stride xBatch). One batch per launch.
// Both tiles staged pair-packed; every fragment dword = one ds_load_b32.
// ---------------------------------------------------------------------------
__global__ __launch_bounds__(256) void gram_pd(
    const bhalf* __restrict__ X, long xBatch, int b, int C,
    const float* __restrict__ xx, float* __restrict__ pd, int N) {
  __shared__ __align__(16) unsigned int As2[16 * 64];  // [kpair][n] pair-packed
  __shared__ __align__(16) unsigned int Bs2[16 * 64];  // [kpair][m] pair-packed
  const int tid = threadIdx.x;
  const int wave = tid >> 5;
  const int lane = tid & 31;
  const int blockM = blockIdx.x * 64;
  const int blockN = blockIdx.y * 64;
  const bhalf* Xb = X + (long)b * xBatch;
  const int nrow = (wave & 3) * 16;
  const int mbase = (wave >> 2) * 32;
  const int r = lane & 15;
  const int half = lane >> 4;
  const int kp = tid >> 4;              // 0..15
  const int pm = (tid & 15) * 4;        // 0..60

  v8f acc0 = {0, 0, 0, 0, 0, 0, 0, 0};
  v8f acc1 = {0, 0, 0, 0, 0, 0, 0, 0};

  for (int k0 = 0; k0 < C; k0 += 32) {
    __syncthreads();
    {
      uint2 ev = *reinterpret_cast<const uint2*>(Xb + (long)(k0 + 2 * kp) * N + blockN + pm);
      uint2 od = *reinterpret_cast<const uint2*>(Xb + (long)(k0 + 2 * kp + 1) * N + blockN + pm);
      uint4 o;
      pack2(ev.x, od.x, o.x, o.y);
      pack2(ev.y, od.y, o.z, o.w);
      *reinterpret_cast<uint4*>(&As2[kp * 64 + pm]) = o;
    }
    {
      uint2 ev = *reinterpret_cast<const uint2*>(Xb + (long)(k0 + 2 * kp) * N + blockM + pm);
      uint2 od = *reinterpret_cast<const uint2*>(Xb + (long)(k0 + 2 * kp + 1) * N + blockM + pm);
      uint4 o;
      pack2(ev.x, od.x, o.x, o.y);
      pack2(ev.y, od.y, o.z, o.w);
      *reinterpret_cast<uint4*>(&Bs2[kp * 64 + pm]) = o;
    }
    __syncthreads();

    union { unsigned int u[8]; v16bf v; } fa, fb0, fb1;
#pragma unroll
    for (int j = 0; j < 8; ++j) {
      int kpa = half * 8 + j;  // A-layout pair index
      fa.u[j] = As2[kpa * 64 + nrow + r];
      int kp2 = half * 8 + j;                          // B-layout pair index
      fb0.u[j] = Bs2[kp2 * 64 + mbase + r];
      fb1.u[j] = Bs2[kp2 * 64 + mbase + 16 + r];
    }
    acc0 = __builtin_amdgcn_wmma_f32_16x16x32_bf16(false, fa.v, false, fb0.v, (short)0, acc0, false, false);
    acc1 = __builtin_amdgcn_wmma_f32_16x16x32_bf16(false, fa.v, false, fb1.v, (short)0, acc1, false, false);
  }
  const float* xxb = xx + (long)b * N;
#pragma unroll
  for (int j = 0; j < 8; ++j) {
    int ng = blockN + nrow + j + 8 * half;
    int mg = blockM + mbase + r;
    float xn = xxb[ng];
    pd[(long)ng * N + mg]      = 2.0f * acc0[j] - xn - xxb[mg];
    pd[(long)ng * N + mg + 16] = 2.0f * acc1[j] - xn - xxb[mg + 16];
  }
}

// ---------------------------------------------------------------------------
// Per-row top-K of pd (one block per row): 128 threads local top-K + merge
// ---------------------------------------------------------------------------
template <int KS>
__global__ __launch_bounds__(128) void topk_rows(const float* __restrict__ pd, int N, int b,
                                                 int* __restrict__ idxOut) {
  const int n = blockIdx.x;
  const int t = threadIdx.x;
  float bv[KS]; int bi[KS];
#pragma unroll
  for (int q = 0; q < KS; ++q) { bv[q] = -3.0e38f; bi[q] = 0; }
  const float* row = pd + (long)n * N;
  for (int m = t; m < N; m += 128) {
    float v = row[m];
    if (v > bv[KS - 1]) {
      float cv = v; int ci = m;
#pragma unroll
      for (int q = 0; q < KS; ++q)
        if (cv > bv[q]) { float tv = bv[q]; int ti = bi[q]; bv[q] = cv; bi[q] = ci; cv = tv; ci = ti; }
    }
  }
  __shared__ float sv[128 * KS];
  __shared__ int   si[128 * KS];
#pragma unroll
  for (int q = 0; q < KS; ++q) { sv[t * KS + q] = bv[q]; si[t * KS + q] = bi[q]; }
  __syncthreads();
  if (t == 0) {
    for (int src = 1; src < 128; ++src) {
      for (int q = 0; q < KS; ++q) {
        float v = sv[src * KS + q];
        if (v <= bv[KS - 1]) break;  // lists are sorted descending
        float cv = v; int ci = si[src * KS + q];
#pragma unroll
        for (int p = 0; p < KS; ++p)
          if (cv > bv[p]) { float tv = bv[p]; int ti = bi[p]; bv[p] = cv; bi[p] = ci; cv = tv; ci = ti; }
      }
    }
    int* op = idxOut + ((long)b * N + n) * KS;
#pragma unroll
    for (int q = 0; q < KS; ++q) op[q] = bi[q];
  }
}

// ---------------------------------------------------------------------------
// sum of squares over 128 channels (bf16 in, f32 out)
// ---------------------------------------------------------------------------
__global__ __launch_bounds__(256) void sqnorm128(const bhalf* __restrict__ X, long xBatch,
                                                 float* __restrict__ xx) {
  const int N = NPTS;
  int n = blockIdx.x * 256 + threadIdx.x;
  int b = blockIdx.z;
  const bhalf* xp = X + (long)b * xBatch + n;
  float s = 0.f;
  for (int c = 0; c < 128; ++c) { float v = b2f(xp[(long)c * N]); s += v * v; }
  xx[(long)b * N + n] = s;
}

// ---------------------------------------------------------------------------
// edge-feature gather: ef[b][c][n*K+k] = X[c][idx]-X[c][n]; ef[b][c+128][..]=X[c][n]
// ---------------------------------------------------------------------------
__global__ __launch_bounds__(256) void gather_edge(
    const bhalf* __restrict__ X, long xBatch,
    const int* __restrict__ idx, int Kn, bhalf* __restrict__ ef) {
  const int N = NPTS, C = 128;
  int n = blockIdx.x * 256 + threadIdx.x;
  int c = blockIdx.y;
  int b = blockIdx.z;
  float xc = b2f(X[(long)b * xBatch + (long)c * N + n]);
  long M = (long)N * Kn;
  long base = (long)b * (2L * C) * M;
  const int* ip = idx + ((long)b * N + n) * Kn;
  bhalf ctr = f2b(xc);
  for (int k = 0; k < Kn; ++k) {
    int j = ip[k];
    float xn = b2f(X[(long)b * xBatch + (long)c * N + j]);
    ef[base + (long)c * M + (long)n * Kn + k] = f2b(xn - xc);
    ef[base + (long)(c + C) * M + (long)n * Kn + k] = ctr;
  }
}

// ---------------------------------------------------------------------------
// max + mean pooling over neighbor axis: in [b][128][n*Kn+k] -> out [b][256][n]
// ---------------------------------------------------------------------------
__global__ __launch_bounds__(256) void pool_maxmean(const bhalf* __restrict__ Yin, int Kn,
                                                    bhalf* __restrict__ pooled) {
  const int N = NPTS;
  int n = blockIdx.x * 256 + threadIdx.x;
  int o = blockIdx.y;
  int b = blockIdx.z;
  const bhalf* src = Yin + ((long)(b * 128 + o) * N) * Kn + (long)n * Kn;
  float mx = -3.0e38f, s = 0.f;
  for (int k = 0; k < Kn; ++k) { float v = b2f(src[k]); mx = fmaxf(mx, v); s += v; }
  pooled[((long)b * 256 + o) * N + n] = f2b(mx);
  pooled[((long)b * 256 + 128 + o) * N + n] = f2b(s / (float)Kn);
}

// ---------------------------------------------------------------------------
// small VALU conv (bf16 in, f32 out)
// ---------------------------------------------------------------------------
__global__ __launch_bounds__(256) void conv_small(
    const bhalf* __restrict__ X, long xBatch, int C,
    const float* __restrict__ W, const float* __restrict__ bias,
    int act, float* __restrict__ Y, long yBatch) {
  const int N = NPTS;
  int n = blockIdx.x * 256 + threadIdx.x;
  int o = blockIdx.y;
  int b = blockIdx.z;
  const bhalf* xp = X + (long)b * xBatch + n;
  const float* w = W + (long)o * C;
  float a = bias[o];
  for (int c = 0; c < C; ++c) a += w[c] * b2f(xp[(long)c * N]);
  if (act) a = lrelu(a);
  Y[(long)b * yBatch + (long)o * N + n] = a;
}

// ---------------------------------------------------------------------------
// per-channel max over n
// ---------------------------------------------------------------------------
__global__ __launch_bounds__(256) void rowmax(const bhalf* __restrict__ Xb,
                                              const float* __restrict__ Xf,
                                              long xBatch, int C, float* __restrict__ outv) {
  const int N = NPTS;
  int c = blockIdx.x;
  int b = blockIdx.z;
  long base = (long)b * xBatch + (long)c * N;
  float mx = -3.0e38f;
  for (int n = threadIdx.x; n < N; n += 256)
    mx = fmaxf(mx, Xb ? b2f(Xb[base + n]) : Xf[base + n]);
  __shared__ float red[256];
  red[threadIdx.x] = mx;
  __syncthreads();
  for (int s = 128; s > 0; s >>= 1) {
    if (threadIdx.x < s) red[threadIdx.x] = fmaxf(red[threadIdx.x], red[threadIdx.x + s]);
    __syncthreads();
  }
  if (threadIdx.x == 0) outv[(long)b * C + c] = red[0];
}

__global__ void softmax_att11(const float* __restrict__ amax, float* __restrict__ att) {
  int b = threadIdx.x;
  if (b >= NB) return;
  float mx = -3.0e38f;
  for (int i = 0; i < 11; ++i) mx = fmaxf(mx, amax[b * 11 + i]);
  float e[11], s = 0.f;
  for (int i = 0; i < 11; ++i) { e[i] = __expf(amax[b * 11 + i] - mx); s += e[i]; }
  for (int i = 0; i < 11; ++i) att[b * 11 + i] = e[i] / s;
}

// ---------------------------------------------------------------------------
// build cu (3200 ch, bf16): [expand, nm1, na1, out3, nm2, na2, out5, nm3, na3, out7, out8]
// ---------------------------------------------------------------------------
__global__ __launch_bounds__(256) void build_cu(
    const float* __restrict__ max8,
    const bhalf* __restrict__ p1, const bhalf* __restrict__ p2, const bhalf* __restrict__ p3,
    const bhalf* __restrict__ cat357, const bhalf* __restrict__ out8,
    bhalf* __restrict__ cu) {
  const int N = NPTS;
  int n = blockIdx.x * 256 + threadIdx.x;
  int c = blockIdx.y;
  int b = blockIdx.z;
  bhalf v;
  if (c < 1024)       v = f2b(max8[(long)b * 1024 + c]);
  else if (c < 1280)  v = p1[((long)b * 256 + (c - 1024)) * N + n];
  else if (c < 1408)  v = cat357[((long)b * 384 + (c - 1280)) * N + n];
  else if (c < 1664)  v = p2[((long)b * 256 + (c - 1408)) * N + n];
  else if (c < 1792)  v = cat357[((long)b * 384 + 128 + (c - 1664)) * N + n];
  else if (c < 2048)  v = p3[((long)b * 256 + (c - 1792)) * N + n];
  else if (c < 2176)  v = cat357[((long)b * 384 + 256 + (c - 2048)) * N + n];
  else                v = out8[((long)b * 1024 + (c - 2176)) * N + n];
  cu[((long)b * 3200 + c) * N + n] = v;
}

__global__ __launch_bounds__(256) void build_wcu(
    const bhalf* __restrict__ cu, const float* __restrict__ att, bhalf* __restrict__ wcu) {
  const int N = NPTS;
  int n = blockIdx.x * 256 + threadIdx.x;
  int c = blockIdx.y;
  int b = blockIdx.z;
  int s = (c < 1024) ? 0 : (c < 2176 ? 1 + (c - 1024) / 128 : 10);
  long i = ((long)b * 3200 + c) * N + n;
  wcu[i] = f2b(att[b * 11 + s] * b2f(cu[i]));
}

// ---------------------------------------------------------------------------
// final: uv->3d via affine patch + xyz; normals = normalize(sn + noff)
// out = [uv_3d (B,3,4,N)] ++ [dense_normal (B,3,4,N)]
// ---------------------------------------------------------------------------
__global__ __launch_bounds__(256) void finalize_out(
    const float* __restrict__ x, const float* __restrict__ uvr,
    const float* __restrict__ patr, const float* __restrict__ nofr,
    float* __restrict__ out) {
  const int N = NPTS;
  int n = blockIdx.x * 256 + threadIdx.x;
  int b = blockIdx.z;
  const float* xb = x + (long)b * 6 * N;
  float px = xb[n], py = xb[N + n], pz = xb[2 * N + n];
  float aff[3][3];
#pragma unroll
  for (int i = 0; i < 3; ++i)
#pragma unroll
    for (int j = 0; j < 3; ++j)
      aff[i][j] = patr[((long)b * 9 + i * 3 + j) * N + n];
  float s0 = aff[2][0], s1 = aff[2][1], s2 = aff[2][2];
  float nr = fmaxf(sqrtf(s0 * s0 + s1 * s1 + s2 * s2), 1e-12f);
  s0 /= nr; s1 /= nr; s2 /= nr;
  float* outN = out + (long)NB * 3 * 4 * N;
#pragma unroll
  for (int r = 0; r < 4; ++r) {
    float u = uvr[((long)b * 8 + r * 2 + 0) * N + n];
    float v = uvr[((long)b * 8 + r * 2 + 1) * N + n];
    float q0 = u * aff[0][0] + v * aff[1][0] + px;
    float q1 = u * aff[0][1] + v * aff[1][1] + py;
    float q2 = u * aff[0][2] + v * aff[1][2] + pz;
    out[(((long)b * 3 + 0) * 4 + r) * N + n] = q0;
    out[(((long)b * 3 + 1) * 4 + r) * N + n] = q1;
    out[(((long)b * 3 + 2) * 4 + r) * N + n] = q2;
    float o0 = nofr[((long)b * 12 + r * 3 + 0) * N + n] + s0;
    float o1 = nofr[((long)b * 12 + r * 3 + 1) * N + n] + s1;
    float o2 = nofr[((long)b * 12 + r * 3 + 2) * N + n] + s2;
    float nn = fmaxf(sqrtf(o0 * o0 + o1 * o1 + o2 * o2), 1e-12f);
    outN[(((long)b * 3 + 0) * 4 + r) * N + n] = o0 / nn;
    outN[(((long)b * 3 + 1) * 4 + r) * N + n] = o1 / nn;
    outN[(((long)b * 3 + 2) * 4 + r) * N + n] = o2 / nn;
  }
}

// ===========================================================================
extern "C" void kernel_launch(void* const* d_in, const int* in_sizes, int n_in,
                              void* d_out, int out_size, void* d_ws, size_t ws_size,
                              hipStream_t stream) {
  (void)in_sizes; (void)n_in; (void)out_size; (void)ws_size;
  const int N = NPTS;
  const float* x = (const float*)d_in[0];

  // layer order: d1..d8, a1, a2, a3, cc, uv, patch, noff  (insertion order)
  static const int LO[15]  = {128, 128, 128, 128, 128, 128, 128, 1024, 256, 128, 11, 256, 8, 9, 12};
  static const int LC[15]  = {6, 128, 256, 256, 256, 256, 256, 384, 3200, 256, 128, 3200, 256, 256, 256};
  static const int LBN[15] = {1, 1, 1, 1, 1, 1, 1, 1, 1, 1, 1, 1, 0, 0, 0};

  const float *Wp[15], *bp[15], *gp[15], *bep[15];
  int ii = 1;
  for (int l = 0; l < 15; ++l) {
    Wp[l] = (const float*)d_in[ii++];
    bp[l] = (const float*)d_in[ii++];
    if (LBN[l]) { gp[l] = (const float*)d_in[ii++]; bep[l] = (const float*)d_in[ii++]; }
    else        { gp[l] = nullptr;                  bep[l] = nullptr; }
  }

  char* base = (char*)d_ws;
  size_t off = 0;
  auto alloc = [&](size_t bytes) -> void* {
    off = (off + 255) & ~(size_t)255;
    void* p = base + off;
    off += bytes;
    return p;
  };

  bhalf* Wb[15]; float* Wf[15]; float* bfold[15];
  for (int l = 0; l < 15; ++l) {
    size_t wc = (size_t)LO[l] * LC[l];
    Wb[l] = (bhalf*)alloc(wc * 2);
    Wf[l] = (float*)alloc(wc * 4);
    bfold[l] = (float*)alloc((size_t)LO[l] * 4);
  }

  int* idx1 = (int*)alloc((size_t)NB * N * 15 * 4);
  int* idx2 = (int*)alloc((size_t)NB * N * 15 * 4);
  int* idx3 = (int*)alloc((size_t)NB * N * 20 * 4);
  bhalf* EB0 = (bhalf*)alloc((size_t)NB * 256 * (size_t)N * 20 * 2);  // edge/cu buffer
  bhalf* EB1 = (bhalf*)alloc((size_t)NB * 3200 * (size_t)N * 2);     // conv-out/wcu buffer
  bhalf* pooled1 = (bhalf*)alloc((size_t)NB * 256 * N * 2);
  bhalf* pooled2 = (bhalf*)alloc((size_t)NB * 256 * N * 2);
  bhalf* pooled3 = (bhalf*)alloc((size_t)NB * 256 * N * 2);
  bhalf* cat357  = (bhalf*)alloc((size_t)NB * 384 * N * 2);          // [out3|out5|out7]
  bhalf* out8    = (bhalf*)alloc((size_t)NB * 1024 * N * 2);
  float* max8    = (float*)alloc((size_t)NB * 1024 * 4);
  bhalf* a1out   = (bhalf*)alloc((size_t)NB * 256 * N * 2);
  bhalf* a2out   = (bhalf*)alloc((size_t)NB * 128 * N * 2);
  float* a3out   = (float*)alloc((size_t)NB * 11 * N * 4);
  float* amax    = (float*)alloc((size_t)NB * 11 * 4);
  float* att     = (float*)alloc((size_t)NB * 11 * 4);
  bhalf* ccout   = (bhalf*)alloc((size_t)NB * 256 * N * 2);
  float* uvr     = (float*)alloc((size_t)NB * 8 * N * 4);
  float* patr    = (float*)alloc((size_t)NB * 9 * N * 4);
  float* nofr    = (float*)alloc((size_t)NB * 12 * N * 4);
  float* xx      = (float*)alloc((size_t)NB * N * 4);
  float* pd      = (float*)alloc((size_t)N * N * 4);

  // 1) fold params
  for (int l = 0; l < 15; ++l) {
    int total = LO[l] * LC[l];
    fold_params<<<dim3((total + 255) / 256, 1, 1), 256, 0, stream>>>(
        Wp[l], bp[l], gp[l], bep[l], LO[l], LC[l], LBN[l], Wb[l], Wf[l], bfold[l]);
  }

  auto gemm = [&](int l, const bhalf* X, long xB, bhalf* Y, long yB, int M, int act) {
    dim3 g(M / 64, LO[l] / 64, NB);
    gemm_bf16<<<g, 256, 0, stream>>>(Wb[l], bfold[l], X, xB, Y, yB, LO[l], LC[l], M, act);
  };

  const int M15 = N * 15, M20 = N * 20;

  // 2) knn on xyz -> idx1
  knn_xyz_k15<<<dim3(N / 256, 1, NB), 256, 0, stream>>>(x, idx1);
  // 3) fused edge + d1 -> out1 (EB0)
  edge_conv_d1<<<dim3(M15 / 256, 1, NB), 256, 0, stream>>>(x, idx1, Wf[0], bfold[0], EB0);
  // 4) d2 -> out2 (EB1)
  gemm(1, EB0, 128L * M15, EB1, 128L * M15, M15, 1);
  // 5) pool1
  pool_maxmean<<<dim3(N / 256, 128, NB), 256, 0, stream>>>(EB1, 15, pooled1);
  // 6) d3 -> out3 (cat357 ch 0..127)
  gemm(2, pooled1, 256L * N, cat357, 384L * N, N, 1);
  // 7-8) knn on out3 -> idx2
  sqnorm128<<<dim3(N / 256, 1, NB), 256, 0, stream>>>(cat357, 384L * N, xx);
  for (int b = 0; b < NB; ++b) {
    gram_pd<<<dim3(64, 64, 1), 256, 0, stream>>>(cat357, 384L * N, b, 128, xx, pd, N);
    topk_rows<15><<<dim3(N, 1, 1), 128, 0, stream>>>(pd, N, b, idx2);
  }
  // 9) gather ef2 (256 ch) -> EB0
  gather_edge<<<dim3(N / 256, 128, NB), 256, 0, stream>>>(cat357, 384L * N, idx2, 15, EB0);
  // 10) d4 -> out4 (EB1)
  gemm(3, EB0, 256L * M15, EB1, 128L * M15, M15, 1);
  // 11) pool2
  pool_maxmean<<<dim3(N / 256, 128, NB), 256, 0, stream>>>(EB1, 15, pooled2);
  // 12) d5 -> out5 (cat357 ch 128..255)
  gemm(4, pooled2, 256L * N, cat357 + 128L * N, 384L * N, N, 1);
  // 13-14) knn on out5 (k=20) -> idx3
  sqnorm128<<<dim3(N / 256, 1, NB), 256, 0, stream>>>(cat357 + 128L * N, 384L * N, xx);
  for (int b = 0; b < NB; ++b) {
    gram_pd<<<dim3(64, 64, 1), 256, 0, stream>>>(cat357 + 128L * N, 384L * N, b, 128, xx, pd, N);
    topk_rows<20><<<dim3(N, 1, 1), 128, 0, stream>>>(pd, N, b, idx3);
  }
  // 15) gather ef3 -> EB0
  gather_edge<<<dim3(N / 256, 128, NB), 256, 0, stream>>>(cat357 + 128L * N, 384L * N, idx3, 20, EB0);
  // 16) d6 -> out6 (EB1)
  gemm(5, EB0, 256L * M20, EB1, 128L * M20, M20, 1);
  // 17) pool3
  pool_maxmean<<<dim3(N / 256, 128, NB), 256, 0, stream>>>(EB1, 20, pooled3);
  // 18) d7 -> out7 (cat357 ch 256..383)
  gemm(6, pooled3, 256L * N, cat357 + 256L * N, 384L * N, N, 1);
  // 19) d8 -> out8
  gemm(7, cat357, 384L * N, out8, 1024L * N, N, 1);
  // 20) expand = max over n of out8
  rowmax<<<dim3(1024, 1, NB), 256, 0, stream>>>(out8, nullptr, 1024L * N, 1024, max8);
  // 21) cu (EB0)
  build_cu<<<dim3(N / 256, 3200, NB), 256, 0, stream>>>(max8, pooled1, pooled2, pooled3,
                                                        cat357, out8, EB0);
  // 22-24) attention trunk
  gemm(8, EB0, 3200L * N, a1out, 256L * N, N, 1);
  gemm(9, a1out, 256L * N, a2out, 128L * N, N, 1);
  conv_small<<<dim3(N / 256, 11, NB), 256, 0, stream>>>(a2out, 128L * N, 128, Wf[10], bfold[10],
                                                        1, a3out, 11L * N);
  rowmax<<<dim3(11, 1, NB), 256, 0, stream>>>(nullptr, a3out, 11L * N, 11, amax);
  softmax_att11<<<1, 32, 0, stream>>>(amax, att);
  // 25) wcu (EB1)
  build_wcu<<<dim3(N / 256, 3200, NB), 256, 0, stream>>>(EB0, att, EB1);
  // 26) cc -> concat features
  gemm(11, EB1, 3200L * N, ccout, 256L * N, N, 1);
  // 27) heads
  conv_small<<<dim3(N / 256, 8, NB), 256, 0, stream>>>(ccout, 256L * N, 256, Wf[12], bfold[12],
                                                       0, uvr, 8L * N);
  conv_small<<<dim3(N / 256, 9, NB), 256, 0, stream>>>(ccout, 256L * N, 256, Wf[13], bfold[13],
                                                       0, patr, 9L * N);
  conv_small<<<dim3(N / 256, 12, NB), 256, 0, stream>>>(ccout, 256L * N, 256, Wf[14], bfold[14],
                                                        0, nofr, 12L * N);
  // 28) finalize
  finalize_out<<<dim3(N / 256, 1, NB), 256, 0, stream>>>(x, uvr, patr, nofr, (float*)d_out);
}
